// QModule_81681688035506
// MI455X (gfx1250) — compile-verified, matrix operand order checked by reading, not run on hardware
//
#include <hip/hip_runtime.h>
#include <hip/hip_bf16.h>

typedef __attribute__((ext_vector_type(8))) int v8i;

#define QMAX 127.0f
#define TM 128                      // block tile M
#define TN 256                      // block tile N
#define TK 64                       // K per LDS tile
#define LSTR 80                     // padded LDS row stride (bytes): 16B aligned, bank-spread
#define ASZ (TM * LSTR)             // A tile bytes in LDS
#define BUFSZ ((TM + TN) * LSTR)    // one double-buffer slot: A tile then B tile

__device__ __forceinline__ int quant1(float v, float zp, float inv) {
    float t = (v - zp) * inv;
    t = fminf(fmaxf(t, -QMAX), QMAX);
    return (int)rintf(t);   // round-half-even == jnp.round
}

// ---- CDNA5 async global->LDS copy (ASYNCcnt-tracked DMA, no VGPR round trip) ----
__device__ __forceinline__ void async_copy_b128(unsigned lds_off, unsigned long long gaddr) {
    asm volatile("global_load_async_to_lds_b128 %0, %1, off"
                 :: "v"(lds_off), "v"(gaddr) : "memory");
}
#define WAIT_ASYNCCNT(n) asm volatile("s_wait_asynccnt %0" :: "i"(n) : "memory")

// ---------------- quantize activations: fp32 -> int8 ----------------
__global__ __launch_bounds__(256) void quant_x_kernel(
    const float* __restrict__ x, const float* __restrict__ ascale,
    const float* __restrict__ azp, signed char* __restrict__ xq, long n)
{
    long i = ((long)blockIdx.x * 256 + threadIdx.x) * 4;
    if (i >= n) return;
    const float inv = 1.0f / ascale[0];
    const float zp  = azp[0];
    float4 v = *(const float4*)(x + i);
    int q0 = quant1(v.x, zp, inv);
    int q1 = quant1(v.y, zp, inv);
    int q2 = quant1(v.z, zp, inv);
    int q3 = quant1(v.w, zp, inv);
    int packed = (q0 & 255) | ((q1 & 255) << 8) | ((q2 & 255) << 16) | (q3 << 24);
    *(int*)(xq + i) = packed;
}

// ---------------- quantize weights (per-row scale): fp32 -> int8 ----------------
__global__ __launch_bounds__(256) void quant_w_kernel(
    const float* __restrict__ w, const float* __restrict__ wscale,
    signed char* __restrict__ wq, int K)
{
    const int row = blockIdx.x;
    const float inv = 1.0f / wscale[row];
    const int c = threadIdx.x * 8;
    const float* p = w + (size_t)row * K + c;
    float4 v0 = *(const float4*)p;
    float4 v1 = *(const float4*)(p + 4);
    int q0 = quant1(v0.x, 0.f, inv), q1 = quant1(v0.y, 0.f, inv);
    int q2 = quant1(v0.z, 0.f, inv), q3 = quant1(v0.w, 0.f, inv);
    int q4 = quant1(v1.x, 0.f, inv), q5 = quant1(v1.y, 0.f, inv);
    int q6 = quant1(v1.z, 0.f, inv), q7 = quant1(v1.w, 0.f, inv);
    int lo = (q0 & 255) | ((q1 & 255) << 8) | ((q2 & 255) << 16) | (q3 << 24);
    int hi = (q4 & 255) | ((q5 & 255) << 8) | ((q6 & 255) << 16) | (q7 << 24);
    int2 pk; pk.x = lo; pk.y = hi;
    *(int2*)(wq + (size_t)row * K + c) = pk;
}

// ---- one K-step of WMMAs for a 64x64 wave tile (4x4 16x16 subtiles) ----
__device__ __forceinline__ void compute_step(
    const signed char* __restrict__ bufA, const signed char* __restrict__ bufB,
    int waveM, int waveN, int fr, int kh, v8i acc[4][4])
{
    // 8-bit A/B fragment layout: lane holds row/col fr, K bytes at
    // kh + {0..7, 16..23, 32..39, 48..55}
    v8i af[4];
    #pragma unroll
    for (int i = 0; i < 4; ++i) {
        const signed char* p = bufA + (waveM * 64 + i * 16 + fr) * LSTR + kh;
        int2 d0 = *(const int2*)(p);
        int2 d1 = *(const int2*)(p + 16);
        int2 d2 = *(const int2*)(p + 32);
        int2 d3 = *(const int2*)(p + 48);
        v8i a; a[0]=d0.x; a[1]=d0.y; a[2]=d1.x; a[3]=d1.y;
               a[4]=d2.x; a[5]=d2.y; a[6]=d3.x; a[7]=d3.y;
        af[i] = a;
    }
    v8i bf[4];
    #pragma unroll
    for (int j = 0; j < 4; ++j) {
        const signed char* p = bufB + (waveN * 64 + j * 16 + fr) * LSTR + kh;
        int2 d0 = *(const int2*)(p);
        int2 d1 = *(const int2*)(p + 16);
        int2 d2 = *(const int2*)(p + 32);
        int2 d3 = *(const int2*)(p + 48);
        v8i b; b[0]=d0.x; b[1]=d0.y; b[2]=d1.x; b[3]=d1.y;
               b[4]=d2.x; b[5]=d2.y; b[6]=d3.x; b[7]=d3.y;
        bf[j] = b;
    }
    #pragma unroll
    for (int i = 0; i < 4; ++i)
        #pragma unroll
        for (int j = 0; j < 4; ++j)
            acc[i][j] = __builtin_amdgcn_wmma_i32_16x16x64_iu8(
                /*sgn_a=*/true, af[i], /*sgn_b=*/true, bf[j],
                acc[i][j], /*reuse_a=*/false, /*reuse_b=*/false);
}

// ---------------- int8 GEMM via V_WMMA_I32_16X16X64_IU8, async double-buffered ----------------
// Aq: [M][K] int8 row-major (x quantized)
// Bq: [N][K] int8 row-major (weight quantized; K-contiguous per output col == B col-major)
__global__ __launch_bounds__(256) void gemm_iu8_kernel(
    const signed char* __restrict__ Aq, const signed char* __restrict__ Bq,
    const float* __restrict__ wscale, const float* __restrict__ ascale,
    const float* __restrict__ bias, float* __restrict__ out,
    int M, int N, int K)
{
    __shared__ signed char lds[2 * BUFSZ];   // [buf][A:128 rows | B:256 rows] x LSTR

    const int tid   = threadIdx.x;
    const int lane  = tid & 31;
    const int wave  = tid >> 5;        // 0..7
    const int waveM = wave >> 2;       // 0..1 -> 64 rows
    const int waveN = wave & 3;        // 0..3 -> 64 cols
    const size_t mBase = (size_t)blockIdx.y * TM;
    const size_t nBase = (size_t)blockIdx.x * TN;

    const int fr = lane & 15;          // row (A) / col (B) within 16
    const int kh = (lane >> 4) << 3;   // K-half offset: 0 or 8 (also M offset in C/D)

    // staging: A tile 128x64 -> 2 b128/thread; B tile 256x64 -> 4 b128/thread
    const int ldrA = tid >> 1;          // 0..127
    const int ldcA = (tid & 1) << 5;    // 0 or 32
    unsigned long long ga = (unsigned long long)(Aq + (mBase + ldrA) * (size_t)K + ldcA);
    unsigned long long gb = (unsigned long long)(Bq + (nBase + tid) * (size_t)K);
    // low 32 bits of a generic pointer to LDS == wave-relative LDS byte offset
    const unsigned ldsBase = (unsigned)(unsigned long long)(void*)lds;
    const unsigned sAoff = ldsBase + ldrA * LSTR + ldcA;
    const unsigned sBoff = ldsBase + ASZ + tid * LSTR;

    v8i acc[4][4];
    #pragma unroll
    for (int i = 0; i < 4; ++i)
        #pragma unroll
        for (int j = 0; j < 4; ++j)
            acc[i][j] = (v8i)0;

    const int nK = K / TK;   // >= 2

    // prologue: fill buffer 0 (6 async b128 per thread)
    async_copy_b128(sAoff,      ga);
    async_copy_b128(sAoff + 16, ga + 16);
    async_copy_b128(sBoff,      gb);
    async_copy_b128(sBoff + 16, gb + 16);
    async_copy_b128(sBoff + 32, gb + 32);
    async_copy_b128(sBoff + 48, gb + 48);

    for (int it = 0; it < nK - 1; ++it) {
        // issue next tile into the other buffer while current computes
        ga += TK; gb += TK;
        const unsigned boff = (unsigned)(((it + 1) & 1) * BUFSZ);
        async_copy_b128(sAoff + boff,      ga);
        async_copy_b128(sAoff + boff + 16, ga + 16);
        async_copy_b128(sBoff + boff,      gb);
        async_copy_b128(sBoff + boff + 16, gb + 16);
        async_copy_b128(sBoff + boff + 32, gb + 32);
        async_copy_b128(sBoff + boff + 48, gb + 48);

        WAIT_ASYNCCNT(6);        // oldest 6 (current buffer) complete; newest 6 in flight
        __syncthreads();         // all waves' current-buffer LDS writes visible

        const signed char* bufA = lds + (it & 1) * BUFSZ;
        compute_step(bufA, bufA + ASZ, waveM, waveN, fr, kh, acc);

        __syncthreads();         // all waves done reading cur before refill (iter it+1)
    }

    // peeled last iteration: nothing left to prefetch
    WAIT_ASYNCCNT(0);
    __syncthreads();
    {
        const signed char* bufA = lds + ((nK - 1) & 1) * BUFSZ;
        compute_step(bufA, bufA + ASZ, waveM, waveN, fr, kh, acc);
    }

    // Epilogue: C/D 16x16 i32 layout -> VGPR r: lanes 0-15 M=r, lanes 16-31 M=r+8
    const float as = ascale[0];
    #pragma unroll
    for (int j = 0; j < 4; ++j) {
        const size_t col = nBase + waveN * 64 + j * 16 + fr;
        const float so = as * wscale[col];
        const float bv = bias[col];
        #pragma unroll
        for (int i = 0; i < 4; ++i) {
            const size_t row = mBase + waveM * 64 + i * 16 + kh;
            float* op = out + row * (size_t)N + col;
            #pragma unroll
            for (int r = 0; r < 8; ++r)
                op[(size_t)r * N] = (float)acc[i][j][r] * so + bv;
        }
    }
}

extern "C" void kernel_launch(void* const* d_in, const int* in_sizes, int n_in,
                              void* d_out, int out_size, void* d_ws, size_t ws_size,
                              hipStream_t stream) {
    const float* x    = (const float*)d_in[0];
    const float* w    = (const float*)d_in[1];
    const float* ws   = (const float*)d_in[2];
    const float* as   = (const float*)d_in[3];
    const float* zp   = (const float*)d_in[4];
    const float* bias = (const float*)d_in[5];
    float* out = (float*)d_out;

    const int B = 4, S = 4096, IN = 2048, OUT = 2048;
    const int M = B * S, N = OUT, K = IN;

    signed char* xq = (signed char*)d_ws;                  // M*K int8 = 32 MB
    signed char* wq = xq + (size_t)M * K;                  // N*K int8 =  4 MB

    const long nx = (long)M * K;
    quant_x_kernel<<<dim3((unsigned)(nx / 4 / 256)), 256, 0, stream>>>(x, as, zp, xq, nx);
    quant_w_kernel<<<dim3(N), 256, 0, stream>>>(w, ws, wq, K);
    gemm_iu8_kernel<<<dim3(N / TN, M / TM), 256, 0, stream>>>(xq, wq, ws, as, bias, out,
                                                              M, N, K);
}
